// RNN_75746043232864
// MI455X (gfx1250) — matrix operand order repaired and checked by reading, not compiled, not run on hardware
//
#include <hip/hip_runtime.h>
#include <hip/hip_bf16.h>

typedef __attribute__((ext_vector_type(16))) _Float16     v16h;
typedef __attribute__((ext_vector_type(8)))  _Float16     v8h;
typedef __attribute__((ext_vector_type(8)))  float        v8f;
typedef __attribute__((ext_vector_type(8)))  unsigned int v8u;

#define B_ 256
#define T_ 2048
#define I_ 25
#define H_ 32

__device__ __forceinline__ float fast_tanh(float x) {
#if __has_builtin(__builtin_amdgcn_tanhf)
    return __builtin_amdgcn_tanhf(x);          // v_tanh_f32 (gfx1250)
#elif __has_builtin(__builtin_amdgcn_tanh_f32)
    return __builtin_amdgcn_tanh_f32(x);
#else
    return tanhf(x);
#endif
}

// Kernel 1: xp[t][b][h] = sum_i x[b][t][i]*W_ih[h][i] + b_ih[h] + b_hh[h]
template <bool XF32>
__global__ __launch_bounds__(256) void rnn_proj(
    const float* __restrict__ x, const float* __restrict__ Wih,
    const float* __restrict__ bih, const float* __restrict__ bhh,
    void* __restrict__ xp)
{
    __shared__ float sW[I_ * H_];
    const int tid = threadIdx.x;
    for (int i = tid; i < I_ * H_; i += 256) sW[i] = Wih[i];
    __syncthreads();

    const size_t g  = (size_t)blockIdx.x * 256 + tid;   // index into [T][B][H]
    const int    h  = (int)(g & (H_ - 1));
    const size_t bt = g >> 5;                           // t*B + b
    const int    b  = (int)(bt & (B_ - 1));
    const int    t  = (int)(bt >> 8);

    float acc = bih[h] + bhh[h];
    const float* xr = x + ((size_t)b * T_ + t) * I_;    // same row for all 32 lanes
    const float* wr = sW + h * I_;
#pragma unroll
    for (int i = 0; i < I_; ++i) acc = fmaf(xr[i], wr[i], acc);
    if constexpr (XF32) ((float*)xp)[g] = acc;
    else                ((_Float16*)xp)[g] = (_Float16)acc;
}

// Kernel 2: WMMA scan, transposed state, row-permuted A tiles.
//   Recurrence per lane: v_wmma -> v_tanh -> v_cvt_pk_rtz -> v_wmma, in-lane.
//   FCOFF: packed f16 state streamed to ws (2x b128 stores), fc done later.
//   else : fc fused, branchless store (hi-lanes dump to own hid slot).
template <bool XF32, bool FCOFF>
__global__ __launch_bounds__(32) void rnn_scan(
    const void* __restrict__ xpv, const float* __restrict__ h0,
    const float* __restrict__ Whh, const float* __restrict__ fcW,
    const float* __restrict__ fcb, float* __restrict__ out,
    unsigned int* __restrict__ hst)
{
    const int lane = threadIdx.x;      // 0..31
    const int col  = lane & 15;        // batch within tile / A-row index m
    const int hi   = lane >> 4;        // half-lane
    const int wav  = blockIdx.x;
    const int b    = wav * 16 + col;

    // Row-permuted A tiles (A layout: lane holds row m = lane&15;
    //   element e -> K = e + 16*hi, this code's k order, same as bs)
    const int r0 = ((col < 8) ? col : col + 8)  * H_;     // rowmap0[m]
    const int r1 = ((col < 8) ? col + 8 : col + 16) * H_; // rowmap1[m]
    v16h a0, a1;
#pragma unroll
    for (int e = 0; e < 8; ++e) {
        a0[e]     = (_Float16)Whh[r0 + e + 16 * hi];
        a0[e + 8] = (_Float16)Whh[r0 + e + 8 + 16 * hi];
        a1[e]     = (_Float16)Whh[r1 + e + 16 * hi];
        a1[e + 8] = (_Float16)Whh[r1 + e + 8 + 16 * hi];
    }

    // B-operand state: bs[e] = h[b][e + 16*hi]
    v16h bs;
#pragma unroll
    for (int e = 0; e < 16; ++e)
        bs[e] = (_Float16)h0[(size_t)b * H_ + e + 16 * hi];

    // fused-fc weights (only used when !FCOFF)
    float fwa[8], fwb[8];
#pragma unroll
    for (int r = 0; r < 8; ++r) {
        fwa[r] = fcW[r + 16 * hi];
        fwb[r] = fcW[r + 8 + 16 * hi];
    }
    const float fcb0 = fcb[0];
    // branchless store targets for fused fc: hi-lanes dump into this wave's own
    // hid slot (overwritten after the loop), so EXEC never changes in the loop.
    float*    obase = (hi == 0) ? (out + (size_t)b * T_)
                                : (out + (size_t)B_ * T_ + (size_t)b * H_);
    const int tmask = (hi == 0) ? ~0 : 0;

    // packed-state stream target (only used when FCOFF): 32B chunk per lane
    unsigned int* hrow = hst + ((size_t)wav * 32 + lane) * 8;

    // xp tile loads, prefetch distance 4 (register ring, constant offsets).
    //   C0 vgpr j -> h = 16*hi + j ; C1 vgpr j -> h = 16*hi + 8 + j
    const _Float16* xrh = (const _Float16*)xpv + (size_t)b * H_;
    const float*    xrf = (const float*)xpv    + (size_t)b * H_;
    v8h qh0[4], qh1[4];
    v8f qf0[4], qf1[4];
#pragma unroll
    for (int u = 0; u < 4; ++u) {
        if constexpr (XF32) {
            qf0[u] = *(const v8f*)(xrf + (size_t)u * (B_ * H_) + 16 * hi);
            qf1[u] = *(const v8f*)(xrf + (size_t)u * (B_ * H_) + 8 + 16 * hi);
        } else {
            qh0[u] = *(const v8h*)(xrh + (size_t)u * (B_ * H_) + 16 * hi);
            qh1[u] = *(const v8h*)(xrh + (size_t)u * (B_ * H_) + 8 + 16 * hi);
        }
    }

    auto step = [&](int t, int u, bool prefetch) {
        v8f c0, c1;
        if constexpr (XF32) { c0 = qf0[u]; c1 = qf1[u]; }
        else {
            const v8h c0h = qh0[u], c1h = qh1[u];
#pragma unroll
            for (int j = 0; j < 8; ++j) { c0[j] = (float)c0h[j]; c1[j] = (float)c1h[j]; }
        }
        if (prefetch) {
            if constexpr (XF32) {
                const float* nx = xrf + (size_t)(t + 4) * (B_ * H_);
                qf0[u] = *(const v8f*)(nx + 16 * hi);
                qf1[u] = *(const v8f*)(nx + 8 + 16 * hi);
            } else {
                const _Float16* nx = xrh + (size_t)(t + 4) * (B_ * H_);
                qh0[u] = *(const v8h*)(nx + 16 * hi);
                qh1[u] = *(const v8h*)(nx + 8 + 16 * hi);
            }
        }

        v8f d0 = __builtin_amdgcn_wmma_f32_16x16x32_f16(false, a0, false, bs,
                                                        (short)0, c0, false, false);
        v8f d1 = __builtin_amdgcn_wmma_f32_16x16x32_f16(false, a1, false, bs,
                                                        (short)0, c1, false, false);

        float t0[8], t1[8];
#pragma unroll
        for (int r = 0; r < 8; ++r) {
            t0[r] = fast_tanh(d0[r]);
            t1[r] = fast_tanh(d1[r]);
        }

        // rebuild B-operand state in-lane: v_cvt_pk_rtz_f16_f32 (critical path)
        v8u bw;
#pragma unroll
        for (int j = 0; j < 4; ++j) {
            bw[j]     = __builtin_bit_cast(unsigned int,
                          __builtin_amdgcn_cvt_pkrtz(t0[2 * j], t0[2 * j + 1]));
            bw[4 + j] = __builtin_bit_cast(unsigned int,
                          __builtin_amdgcn_cvt_pkrtz(t1[2 * j], t1[2 * j + 1]));
        }
        bs = __builtin_bit_cast(v16h, bw);

        if constexpr (FCOFF) {
            // stream packed state; fc handled by rnn_fc kernel
            unsigned int* hp = hrow + (size_t)t * (16 * 32 * 8);
            *(v8u*)hp = bw;                         // 2x global_store_b128
        } else {
            float pa = 0.0f, pb = 0.0f, pc = 0.0f, pd = 0.0f;
#pragma unroll
            for (int r = 0; r < 4; ++r) {
                pa = fmaf(t0[2 * r],     fwa[2 * r],     pa);
                pb = fmaf(t0[2 * r + 1], fwa[2 * r + 1], pb);
                pc = fmaf(t1[2 * r],     fwb[2 * r],     pc);
                pd = fmaf(t1[2 * r + 1], fwb[2 * r + 1], pd);
            }
            float p = (pa + pb) + (pc + pd);
            p += __shfl_xor(p, 16, 32);             // combine the two h-halves
            __builtin_nontemporal_store(p + fcb0, obase + (t & tmask));
        }
    };

    for (int tb = 0; tb < T_ - 4; tb += 4) {        // main loop: unconditional prefetch
#pragma unroll
        for (int u = 0; u < 4; ++u) step(tb + u, u, true);
    }
#pragma unroll
    for (int u = 0; u < 4; ++u) step(T_ - 4 + u, u, false);  // peeled tail

    // final hidden state, d_out tail: [1][B][H] f32
    float* hid = out + (size_t)B_ * T_;
#pragma unroll
    for (int e = 0; e < 16; ++e)
        hid[(size_t)b * H_ + e + 16 * hi] = (float)bs[e];
}

// Kernel 3: out[b][t] = fc_b + sum_h hstate[t][b][h]*fc_W[h]  (memory-bound GEMV)
// hst chunk layout: ((t*16 + b/16)*32 + lane)*32B, lane = b%16 (+16 for h>=16).
__global__ __launch_bounds__(256) void rnn_fc(
    const unsigned int* __restrict__ hst, const float* __restrict__ fcW,
    const float* __restrict__ fcb, float* __restrict__ out)
{
    const size_t g  = (size_t)blockIdx.x * 256 + threadIdx.x;  // 131072 threads
    const int    b  = (int)(g & (B_ - 1));
    const int    t0 = (int)(g >> 8) * 4;
    const size_t cbase = ((size_t)(b >> 4)) * 32 + (b & 15);
    const float  fb = fcb[0];

    float4 o;
#pragma unroll
    for (int u = 0; u < 4; ++u) {
        const int t = t0 + u;
        const v8u* p0 = (const v8u*)hst + ((size_t)t * 16 * 32 + cbase);
        const v16h hA = __builtin_bit_cast(v16h, p0[0]);    // h 0..15
        const v16h hB = __builtin_bit_cast(v16h, p0[16]);   // h 16..31
        float pa = 0.0f, pb = 0.0f;
#pragma unroll
        for (int e = 0; e < 16; ++e) {
            pa = fmaf((float)hA[e], fcW[e],      pa);
            pb = fmaf((float)hB[e], fcW[16 + e], pb);
        }
        o[u == 0 ? 0 : (u == 1 ? 1 : (u == 2 ? 2 : 3))] = fb + pa + pb;
    }
    *(float4*)(out + (size_t)b * T_ + t0) = o;
}

extern "C" void kernel_launch(void* const* d_in, const int* in_sizes, int n_in,
                              void* d_out, int out_size, void* d_ws, size_t ws_size,
                              hipStream_t stream) {
    const float* x   = (const float*)d_in[0];   // [B,T,I]
    const float* h0  = (const float*)d_in[1];   // [1,B,H]
    const float* Wih = (const float*)d_in[2];   // [H,I]
    const float* Whh = (const float*)d_in[3];   // [H,H]
    const float* bih = (const float*)d_in[4];   // [H]
    const float* bhh = (const float*)d_in[5];   // [H]
    const float* fcW = (const float*)d_in[6];   // [O,H]
    const float* fcb = (const float*)d_in[7];   // [O]
    float* out = (float*)d_out;                 // [B,T,O] ++ [1,B,H]
    (void)in_sizes; (void)n_in; (void)out_size;

    const size_t total  = (size_t)T_ * B_ * H_;         // 16,777,216 elements
    const size_t xpF32  = total * 4;                    // 64 MiB
    const size_t xpF16  = total * 2;                    // 32 MiB
    const size_t hstSz  = total * 2;                    // 32 MiB packed state
    const int    nProjB = (int)(total / 256);
    const int    nFcB   = (B_ * T_ / 4) / 256;          // 512 blocks

    if (ws_size >= xpF32 + hstSz) {                     // f32 xp + fc offload
        unsigned int* hst = (unsigned int*)((char*)d_ws + xpF32);
        rnn_proj<true><<<nProjB, 256, 0, stream>>>(x, Wih, bih, bhh, d_ws);
        rnn_scan<true, true><<<B_ / 16, 32, 0, stream>>>(d_ws, h0, Whh, fcW, fcb, out, hst);
        rnn_fc<<<nFcB, 256, 0, stream>>>(hst, fcW, fcb, out);
    } else if (ws_size >= xpF16 + hstSz) {              // f16 xp + fc offload
        unsigned int* hst = (unsigned int*)((char*)d_ws + xpF16);
        rnn_proj<false><<<nProjB, 256, 0, stream>>>(x, Wih, bih, bhh, d_ws);
        rnn_scan<false, true><<<B_ / 16, 32, 0, stream>>>(d_ws, h0, Whh, fcW, fcb, out, hst);
        rnn_fc<<<nFcB, 256, 0, stream>>>(hst, fcW, fcb, out);
    } else {                                            // f16 xp + fused fc
        rnn_proj<false><<<nProjB, 256, 0, stream>>>(x, Wih, bih, bhh, d_ws);
        rnn_scan<false, false><<<B_ / 16, 32, 0, stream>>>(d_ws, h0, Whh, fcW, fcb, out, nullptr);
    }
}
